// CausalSelfAttention_11269994185380
// MI455X (gfx1250) — matrix operand order repaired and checked by reading, not compiled, not run on hardware
//
#include <hip/hip_runtime.h>

typedef __bf16 bf16_t;
typedef __attribute__((ext_vector_type(16))) __bf16 v16bf;
typedef __attribute__((ext_vector_type(8)))  __bf16 v8bf;
typedef __attribute__((ext_vector_type(4)))  __bf16 v4bf;
typedef __attribute__((ext_vector_type(8)))  float   v8f;
typedef __attribute__((ext_vector_type(4)))  float   v4f;

constexpr int N_EMBD = 1024;
constexpr int N_HEAD = 16;
constexpr int HEAD   = 64;
constexpr int BSZ    = 2;
constexpr int SEQ    = 2048;
constexpr int MTOT   = BSZ * SEQ;   // 4096 rows

__device__ __forceinline__ v8f wmma_bf16(v16bf a, v16bf b, v8f c) {
  // (neg_a, A, neg_b, B, c_mod, C, reuse_a, reuse_b)
  return __builtin_amdgcn_wmma_f32_16x16x32_bf16(false, a, false, b, (short)0, c,
                                                 false, false);
}

// Generic pointer to a __shared__ object: low 32 bits are the wave/group LDS
// byte offset (LDS aperture encoding), which is what the async-to-LDS VDST wants.
__device__ __forceinline__ uint32_t lds_off(const void* p) {
  return (uint32_t)(uintptr_t)p;
}

// GLOBAL_LOAD_ASYNC_TO_LDS_B128 (GV mode): 16B global -> LDS, tracked by ASYNCcnt.
__device__ __forceinline__ void async_copy_b128(uint32_t lds_byte_off,
                                                const void* gaddr) {
  asm volatile("global_load_async_to_lds_b128 %0, %1, off"
               :: "v"(lds_byte_off), "v"(gaddr)
               : "memory");
}

__device__ __forceinline__ void wait_async0() {
  asm volatile("s_wait_asynccnt 0x0" ::: "memory");
}

// A fragment: 16(M) x 32(K) tile, row-major, leading dim ld (elements).
// Lane L<16 holds row L, K = {0..7, 16..23}; lane L+16 holds row L, K = {8..15, 24..31}.
__device__ __forceinline__ v16bf load_frag_a(const bf16_t* __restrict__ base, int ld) {
  int lane = threadIdx.x & 31;
  int r = lane & 15, h = lane >> 4;
  const bf16_t* p = base + (size_t)r * ld + h * 8;
  v16bf f;
  *((v8bf*)&f)     = *(const v8bf*)(p);
  *((v8bf*)&f + 1) = *(const v8bf*)(p + 16);
  return f;
}

// B fragment: 32(K) x 16(N); memory holds B^T (N rows x K cols, K contiguous).
// Lane n<16 holds col n, K = {0..7, 16..23}; lane n+16 holds col n, K = {8..15, 24..31}.
__device__ __forceinline__ v16bf load_frag_b(const bf16_t* __restrict__ base, int ld) {
  int lane = threadIdx.x & 31;
  int r = lane & 15, h = lane >> 4;
  const bf16_t* p = base + (size_t)r * ld + h * 8;
  v16bf f;
  *((v8bf*)&f)     = *(const v8bf*)(p);
  *((v8bf*)&f + 1) = *(const v8bf*)(p + 16);
  return f;
}

// ---------------------------------------------------------------- cast ------
__global__ void cast_f32_to_bf16(const float* __restrict__ in,
                                 bf16_t* __restrict__ out, int n4) {
  int i = blockIdx.x * blockDim.x + threadIdx.x;
  if (i >= n4) return;
  v4f v = *(const v4f*)(in + (size_t)i * 4);
  v4bf o;
  o[0] = (bf16_t)v[0]; o[1] = (bf16_t)v[1];
  o[2] = (bf16_t)v[2]; o[3] = (bf16_t)v[3];
  *(v4bf*)(out + (size_t)i * 4) = o;
}

// --------------------------- LDS-staged, async double-buffered WMMA GEMM ----
// C = A(MxK) * W^T(KxN); A,W bf16 (both K-contiguous row-major).
// mode 0: bf16 out[((b*H+h)*T + t)*64 + d]   (head-major, Q/K)
// mode 1: bf16 out[((b*H+h)*64 + d)*T + t]   (head-major transposed, V)
// mode 2: f32  out[m*N_EMBD + col] + bias    (final projection)
__global__ void __launch_bounds__(128) gemm_wmma_staged(
    const bf16_t* __restrict__ A, const bf16_t* __restrict__ W,
    bf16_t* __restrict__ outb, float* __restrict__ outf,
    const float* __restrict__ bias, int mode) {
  constexpr int K = N_EMBD;
  constexpr int KSTEP = 32;
  __shared__ __align__(16) bf16_t lA[2][128 * KSTEP];  // 8KB per buffer
  __shared__ __align__(16) bf16_t lB[2][128 * KSTEP];

  int tid  = threadIdx.x;
  int lane = tid & 31;
  int wid  = tid >> 5;                       // 0..3
  int m0 = blockIdx.y * 128;
  int n0 = blockIdx.x * 128;
  int wm = (wid >> 1) * 64;                  // wave 64x64 tile inside block
  int wn = (wid & 1) * 64;

  // Thread t async-copies row t (64B = 4x b128) of the A block and the B block.
  auto stage = [&](int buf, int k0) {
    const bf16_t* ga = A + (size_t)(m0 + tid) * K + k0;
    const bf16_t* gb = W + (size_t)(n0 + tid) * K + k0;
    uint32_t la = lds_off(&lA[buf][tid * KSTEP]);
    uint32_t lb = lds_off(&lB[buf][tid * KSTEP]);
#pragma unroll
    for (int s = 0; s < 4; ++s) {
      async_copy_b128(la + s * 16, ga + s * 8);
      async_copy_b128(lb + s * 16, gb + s * 8);
    }
  };

  stage(0, 0);
  wait_async0();
  __syncthreads();

  v8f acc[4][4] = {};
  for (int kk = 0; kk < K / KSTEP; ++kk) {
    int buf = kk & 1;
    if (kk + 1 < K / KSTEP) stage(buf ^ 1, (kk + 1) * KSTEP);

    v16bf af[4], bfg[4];
#pragma unroll
    for (int i = 0; i < 4; ++i)
      af[i] = load_frag_a(&lA[buf][(wm + 16 * i) * KSTEP], KSTEP);
#pragma unroll
    for (int j = 0; j < 4; ++j)
      bfg[j] = load_frag_b(&lB[buf][(wn + 16 * j) * KSTEP], KSTEP);
#pragma unroll
    for (int i = 0; i < 4; ++i)
#pragma unroll
      for (int j = 0; j < 4; ++j)
        acc[i][j] = wmma_bf16(af[i], bfg[j], acc[i][j]);

    wait_async0();        // next buffer resident in LDS
    __syncthreads();      // everyone done reading current buffer
  }

  int half = lane >> 4, cl = lane & 15;
#pragma unroll
  for (int i = 0; i < 4; ++i)
#pragma unroll
    for (int j = 0; j < 4; ++j) {
      int col = n0 + wn + j * 16 + cl;
      int hh = col >> 6, dd = col & 63;
      float bv = (mode == 2) ? bias[col] : 0.f;
#pragma unroll
      for (int r = 0; r < 8; ++r) {
        int mrow = m0 + wm + i * 16 + r + 8 * half;
        int bb = mrow >> 11, tt = mrow & (SEQ - 1);
        if (mode == 0)
          outb[(((size_t)bb * N_HEAD + hh) * SEQ + tt) * HEAD + dd] =
              (bf16_t)acc[i][j][r];
        else if (mode == 1)
          outb[(((size_t)bb * N_HEAD + hh) * HEAD + dd) * SEQ + tt] =
              (bf16_t)acc[i][j][r];
        else
          outf[(size_t)mrow * N_EMBD + col] = acc[i][j][r] + bv;
      }
    }
}

// ------------------------------------------------------ flash attention -----
// Q,K: [B,H,T,64] bf16;  Vt: [B,H,64,T] bf16;  ctx: [B*T, 1024] bf16.
// One 16-query tile per wave; barrier-free (divergent causal trip counts).
__global__ void __launch_bounds__(128) flash_attn_bf16(
    const bf16_t* __restrict__ Q, const bf16_t* __restrict__ Kb,
    const bf16_t* __restrict__ Vt, bf16_t* __restrict__ ctx) {
  __shared__ __align__(16) bf16_t lds_p[4][16 * 48];  // 16x32 P tile, padded rows

  int lane = threadIdx.x & 31;
  int w    = threadIdx.x >> 5;
  int bh   = blockIdx.y;                        // b*16 + h
  int q0   = (blockIdx.x * 4 + w) * 16;

  const bf16_t* Qh = Q  + (size_t)bh * SEQ * HEAD;
  const bf16_t* Kh = Kb + (size_t)bh * SEQ * HEAD;
  const bf16_t* Vh = Vt + (size_t)bh * HEAD * SEQ;

  v16bf qa0 = load_frag_a(Qh + (size_t)q0 * HEAD, HEAD);       // d = 0..31
  v16bf qa1 = load_frag_a(Qh + (size_t)q0 * HEAD + 32, HEAD);  // d = 32..63

  v8f o[4] = {};          // 16q x 64d output accumulators
  float rmax[8], rsum[8];
#pragma unroll
  for (int i = 0; i < 8; ++i) { rmax[i] = -__builtin_inff(); rsum[i] = 0.f; }

  int half = lane >> 4, cl = lane & 15;
  const float scale = 0.125f;                   // 1/sqrt(64)
  int ktmax = (q0 + 15) >> 5;                   // last 32-key tile touching diag
  for (int kt = 0; kt <= ktmax; ++kt) {
    int k0 = kt * 32;
    if (kt < ktmax) {                           // prefetch next K tile into L2
      __builtin_prefetch(Kh + (size_t)(k0 + 32) * HEAD + lane * 64, 0, 0);
    }
    // S = Q * K^T  (two 16x16 score tiles over the 32-key block)
    v8f slo = {}, shi = {};
    slo = wmma_bf16(qa0, load_frag_b(Kh + (size_t)k0 * HEAD, HEAD), slo);
    slo = wmma_bf16(qa1, load_frag_b(Kh + (size_t)k0 * HEAD + 32, HEAD), slo);
    shi = wmma_bf16(qa0, load_frag_b(Kh + (size_t)(k0 + 16) * HEAD, HEAD), shi);
    shi = wmma_bf16(qa1, load_frag_b(Kh + (size_t)(k0 + 16) * HEAD + 32, HEAD), shi);

    int keyLo = k0 + cl, keyHi = keyLo + 16;
#pragma unroll
    for (int i = 0; i < 8; ++i) {
      int qrow = q0 + i + 8 * half;
      float sl = (keyLo <= qrow) ? slo[i] * scale : -__builtin_inff();
      float sh = (keyHi <= qrow) ? shi[i] * scale : -__builtin_inff();
      // row max across the 16 lanes of this half-wave
      float tmax = fmaxf(sl, sh);
      tmax = fmaxf(tmax, __shfl_xor(tmax, 1, 32));
      tmax = fmaxf(tmax, __shfl_xor(tmax, 2, 32));
      tmax = fmaxf(tmax, __shfl_xor(tmax, 4, 32));
      tmax = fmaxf(tmax, __shfl_xor(tmax, 8, 32));
      float mnew = fmaxf(rmax[i], tmax);
      float corr = __expf(rmax[i] - mnew);
      float pl = __expf(sl - mnew);
      float ph = __expf(sh - mnew);
      float ts = pl + ph;
      ts += __shfl_xor(ts, 1, 32);
      ts += __shfl_xor(ts, 2, 32);
      ts += __shfl_xor(ts, 4, 32);
      ts += __shfl_xor(ts, 8, 32);
      rsum[i] = rsum[i] * corr + ts;
      rmax[i] = mnew;
      int row = i + 8 * half;
      lds_p[w][row * 48 + cl]      = (bf16_t)pl;
      lds_p[w][row * 48 + 16 + cl] = (bf16_t)ph;
#pragma unroll
      for (int t = 0; t < 4; ++t) o[t][i] = o[t][i] * corr;
    }
    // reload P (16x32 bf16) from LDS in A-fragment layout (intra-wave, in-order LDS)
    v16bf pa;
    {
      const bf16_t* p = &lds_p[w][cl * 48 + half * 8];
      *((v8bf*)&pa)     = *(const v8bf*)(p);
      *((v8bf*)&pa + 1) = *(const v8bf*)(p + 16);
    }
    // O += P * V : Vt rows are d (contiguous over t), exactly B^T layout
#pragma unroll
    for (int t = 0; t < 4; ++t) {
      v16bf vb = load_frag_b(Vh + (size_t)(t * 16) * SEQ + k0, SEQ);
      o[t] = wmma_bf16(pa, vb, o[t]);
    }
  }

  // normalize + store context, row-major [B*T, 1024] for the proj GEMM
  int b = bh >> 4, hh = bh & 15;
#pragma unroll
  for (int i = 0; i < 8; ++i) {
    float inv = 1.0f / rsum[i];
    int qrow = q0 + i + 8 * half;
    size_t rowbase = (size_t)(b * SEQ + qrow) * N_EMBD + hh * HEAD;
#pragma unroll
    for (int t = 0; t < 4; ++t)
      ctx[rowbase + t * 16 + cl] = (bf16_t)(o[t][i] * inv);
  }
}

// ----------------------------------------------------------------- host -----
extern "C" void kernel_launch(void* const* d_in, const int* in_sizes, int n_in,
                              void* d_out, int out_size, void* d_ws, size_t ws_size,
                              hipStream_t stream) {
  (void)in_sizes; (void)n_in; (void)out_size; (void)ws_size;
  const float* x  = (const float*)d_in[0];
  const float* wq = (const float*)d_in[1];
  const float* wk = (const float*)d_in[2];
  const float* wv = (const float*)d_in[3];
  const float* wp = (const float*)d_in[4];
  const float* bp = (const float*)d_in[5];

  const size_t XE = (size_t)MTOT * N_EMBD;    // 4,194,304 (also B*H*T*D)
  const size_t WE = (size_t)N_EMBD * N_EMBD;  // 1,048,576

  char* ws = (char*)d_ws;
  size_t off = 0;
  auto carve = [&](size_t elems) {
    void* p = ws + off;
    off += (elems * sizeof(bf16_t) + 255) & ~(size_t)255;
    return (bf16_t*)p;
  };
  bf16_t* xb   = carve(XE);
  bf16_t* wqb  = carve(WE);
  bf16_t* wkb  = carve(WE);
  bf16_t* wvb  = carve(WE);
  bf16_t* wpb  = carve(WE);
  bf16_t* Qb   = carve(XE);   // [B,H,T,64]
  bf16_t* Kb   = carve(XE);   // [B,H,T,64]
  bf16_t* Vtb  = carve(XE);   // [B,H,64,T]
  bf16_t* ctxb = carve(XE);   // [B*T, 1024]

  // fp32 -> bf16 casts
  cast_f32_to_bf16<<<(int)(XE / 4 / 256), 256, 0, stream>>>(x,  xb,  (int)(XE / 4));
  cast_f32_to_bf16<<<(int)(WE / 4 / 256), 256, 0, stream>>>(wq, wqb, (int)(WE / 4));
  cast_f32_to_bf16<<<(int)(WE / 4 / 256), 256, 0, stream>>>(wk, wkb, (int)(WE / 4));
  cast_f32_to_bf16<<<(int)(WE / 4 / 256), 256, 0, stream>>>(wv, wvb, (int)(WE / 4));
  cast_f32_to_bf16<<<(int)(WE / 4 / 256), 256, 0, stream>>>(wp, wpb, (int)(WE / 4));

  dim3 ggrid(N_EMBD / 128, MTOT / 128);       // (8, 32)
  gemm_wmma_staged<<<ggrid, 128, 0, stream>>>(xb, wqb, Qb,  nullptr, nullptr, 0);
  gemm_wmma_staged<<<ggrid, 128, 0, stream>>>(xb, wkb, Kb,  nullptr, nullptr, 0);
  gemm_wmma_staged<<<ggrid, 128, 0, stream>>>(xb, wvb, Vtb, nullptr, nullptr, 1);

  dim3 fgrid(SEQ / 64, BSZ * N_HEAD);         // (32, 32), 4 waves/block
  flash_attn_bf16<<<fgrid, 128, 0, stream>>>(Qb, Kb, Vtb, ctxb);

  gemm_wmma_staged<<<ggrid, 128, 0, stream>>>(ctxb, wpb, nullptr,
                                              (float*)d_out, bp, 2);
}
